// SpatioTemporalGAT_29669634080936
// MI455X (gfx1250) — matrix operand order repaired and checked by reading, not compiled, not run on hardware
//
#include <hip/hip_runtime.h>
#include <hip/hip_bf16.h>
#include <cstdint>

// Problem constants (match reference)
#define B_   32
#define T_   32
#define N_   256
#define FIN  32
#define FOUT 32
#define H_   256
#define IN_  (N_ * FOUT)       // 8192
#define G4H  (4 * H_)          // 1024
#define NEG_SLOPE 0.2f

typedef __attribute__((ext_vector_type(16))) __bf16 v16bf;
typedef __attribute__((ext_vector_type(8)))  float  v8f;

// ---------------------------------------------------------------------------
// WMMA fragment loaders (wave32); A and B both use contiguous k = 16h+e
// ---------------------------------------------------------------------------

// A-matrix 16x32 bf16, source row-major with leading dim ld (elements).
// lane: m = lane&15, h = lane>>4.  K(e) = 16h+e.
__device__ inline v16bf load_a_rm(const __bf16* __restrict__ p, int ld) {
  const int lane = threadIdx.x & 31;
  const __bf16* r = p + (size_t)(lane & 15) * ld + 16 * (lane >> 4);
  v16bf a;
#pragma unroll
  for (int e = 0; e < 8; ++e) a[e] = r[e];
#pragma unroll
  for (int e = 0; e < 8; ++e) a[8 + e] = r[8 + e];
  return a;
}

// B-matrix 32x16 bf16 stored row-major [k][n] with leading dim ld.
// lane: n = lane&15, kbase = 16*(lane>>4).  element e <-> k = kbase+e.
__device__ inline v16bf load_b_kn(const __bf16* __restrict__ p, int ld) {
  const int lane = threadIdx.x & 31;
  const int n  = lane & 15;
  const int kb = (lane >> 4) * 16;
  v16bf b;
#pragma unroll
  for (int e = 0; e < 16; ++e) b[e] = p[(size_t)(kb + e) * ld + n];
  return b;
}

// B-matrix 32x16 where logical B[k][n] = w[(n0+n)*ldw + k0+k] (w row-major in n,
// contiguous in k -> each lane reads 16 contiguous bf16 = 32 bytes).
__device__ inline v16bf load_b_nk(const __bf16* __restrict__ w, int ldw,
                                  int n0, int k0) {
  const int lane = threadIdx.x & 31;
  const __bf16* r = w + (size_t)(n0 + (lane & 15)) * ldw + k0 + 16 * (lane >> 4);
  v16bf b;
#pragma unroll
  for (int e = 0; e < 16; ++e) b[e] = r[e];
  return b;
}

__device__ inline float sigf(float x) { return 1.0f / (1.0f + __expf(-x)); }

// ---------------------------------------------------------------------------
// Kernel 0: f32 -> bf16 convert (weights, done once per launch)
// ---------------------------------------------------------------------------
__global__ void f32_to_bf16_kernel(const float* __restrict__ in,
                                   __bf16* __restrict__ out, int n) {
  int i = blockIdx.x * blockDim.x + threadIdx.x;
  if (i < n) out[i] = (__bf16)in[i];
}

// ---------------------------------------------------------------------------
// Kernel 1: per-(b,t) GAT. One workgroup (256 thr = 8 waves) per (b,t).
// Output written directly as bf16 rows of the LSTM input matrix X[1024][8192].
// ---------------------------------------------------------------------------
__global__ __launch_bounds__(256) void gat_kernel(
    const float* __restrict__ x, const float* __restrict__ adj,
    const float* __restrict__ Wg, const float* __restrict__ a_src,
    const float* __restrict__ a_dst, const float* __restrict__ gat_b,
    __bf16* __restrict__ Xb) {
  extern __shared__ char smem[];
  // sP (bf16 256x256 = 128KB) aliases sX (f32 256x32 = 32KB): sX is dead
  // before the first sP write (separated by __syncthreads()).
  __bf16* sP  = (__bf16*)smem;                    // 131072 B
  float*  sX  = (float*)smem;                     // alias, 32768 B
  float*  sW  = (float*)(smem + 131072);          // 4096 B
  float*  sAs = sW + FIN * FOUT;                  // 128 B
  float*  sAd = sAs + FOUT;                       // 128 B
  float*  sSrc = sAd + FOUT;                      // 1024 B
  float*  sDst = sSrc + N_;                       // 1024 B
  float*  sInv = sDst + N_;                       // 1024 B
  __bf16* sHb  = (__bf16*)(sInv + N_);            // 16384 B
  // total: 154,880 B

  const int bt  = blockIdx.x;
  const int tid = threadIdx.x;
  const float* xbt   = x   + (size_t)bt * N_ * FIN;
  const float* adjbt = adj + (size_t)bt * N_ * N_;

  // ---- stage x tile, W, attention vectors --------------------------------
  for (int k = tid; k < N_ * FIN; k += 256) sX[k] = xbt[k];
  for (int k = tid; k < FIN * FOUT; k += 256) sW[k] = Wg[k];
  if (tid < FOUT) { sAs[tid] = a_src[tid]; sAd[tid] = a_dst[tid]; }
  __syncthreads();

  // ---- h = x @ W   (one row per thread, f32) -----------------------------
  {
    float xr[FIN];
#pragma unroll
    for (int f = 0; f < FIN; ++f) xr[f] = sX[tid * FIN + f];
    float ss = 0.f, sd = 0.f;
#pragma unroll 4
    for (int o = 0; o < FOUT; ++o) {
      float acc = 0.f;
#pragma unroll
      for (int f = 0; f < FIN; ++f) acc += xr[f] * sW[f * FOUT + o];
      sHb[tid * FOUT + o] = (__bf16)acc;   // bf16 B-operand for WMMA
      ss += acc * sAs[o];
      sd += acc * sAd[o];
    }
    sSrc[tid] = ss;
    sDst[tid] = sd;
  }
  __syncthreads();  // sX now dead -> sP region free

  // ---- masked softmax over sources j, per target row i = tid -------------
  {
    const int i = tid;
    const float sdv = sDst[i];
    float m = -3.4e38f;
    for (int j = 0; j < N_; ++j) {           // coalesced: lane i contiguous
      const float a = adjbt[j * N_ + i];     // adj[b,t,j,i]  (== adj^T[i,j])
      if (a != 0.f || j == i) {
        float e = sdv + sSrc[j];
        e = e >= 0.f ? e : NEG_SLOPE * e;
        m = fmaxf(m, e);
      }
    }
    float l = 0.f;
    for (int j = 0; j < N_; ++j) {           // 2nd pass hits L2 (tile 256KB)
      const float a = adjbt[j * N_ + i];
      float p = 0.f;
      if (a != 0.f || j == i) {
        float e = sdv + sSrc[j];
        e = e >= 0.f ? e : NEG_SLOPE * e;
        p = __expf(e - m);
        l += p;
      }
      sP[i * N_ + j] = (__bf16)p;            // unnormalized; 1/l in epilogue
    }
    sInv[i] = 1.f / l;
  }
  __syncthreads();

  // ---- out = (P @ h) * (1/l) + b  via v_wmma_f32_16x16x32_bf16 -----------
  const int lane = tid & 31;
  const int wave = tid >> 5;
  for (int tIdx = wave; tIdx < 32; tIdx += 8) {   // 16 M-tiles x 2 N-tiles
    const int mi = tIdx >> 1, ni = tIdx & 1;
    v8f acc = {};
#pragma unroll
    for (int kk = 0; kk < 8; ++kk) {              // K = 256 = 8 x 32
      v16bf A  = load_a_rm(sP + (size_t)mi * 16 * N_ + kk * 32, N_);
      v16bf Bv = load_b_kn(sHb + (size_t)kk * 32 * FOUT + ni * 16, FOUT);
      acc = __builtin_amdgcn_wmma_f32_16x16x32_bf16(
          false, A, false, Bv, (short)0, acc, false, false);
    }
    const int n  = ni * 16 + (lane & 15);         // output feature o (0..31)
    const int mb = mi * 16 + (lane >> 4) * 8;     // base node row
    const float gb = gat_b[n];
#pragma unroll
    for (int r = 0; r < 8; ++r) {
      const int mm = mb + r;
      const float v = acc[r] * sInv[mm] + gb;
      Xb[(size_t)bt * IN_ + mm * FOUT + n] = (__bf16)v;
    }
  }
}

// ---------------------------------------------------------------------------
// Kernel 2: G = X @ W_ih^T   (1024 x 8192 x 1024, bf16 WMMA, f32 out)
// WG = 256 thr (8 waves); wave computes a 16x64 strip; WG tile 128x64.
// ---------------------------------------------------------------------------
__global__ __launch_bounds__(256) void lstm_in_gemm(
    const __bf16* __restrict__ Xb, const __bf16* __restrict__ Wihb,
    float* __restrict__ G) {
  const int lane = threadIdx.x & 31;
  const int wave = threadIdx.x >> 5;
  const int m0 = blockIdx.y * 128 + wave * 16;
  const int n0 = blockIdx.x * 64;

  v8f acc[4] = {{}, {}, {}, {}};
  for (int k0 = 0; k0 < IN_; k0 += 32) {
    v16bf A = load_a_rm(Xb + (size_t)m0 * IN_ + k0, IN_);
#pragma unroll
    for (int j = 0; j < 4; ++j) {
      v16bf Bv = load_b_nk(Wihb, IN_, n0 + j * 16, k0);  // B[k][n]=W_ih[n][k]
      acc[j] = __builtin_amdgcn_wmma_f32_16x16x32_bf16(
          false, A, false, Bv, (short)0, acc[j], false, false);
    }
  }
  const int nn = lane & 15;
  const int mb = (lane >> 4) * 8;
#pragma unroll
  for (int j = 0; j < 4; ++j)
#pragma unroll
    for (int r = 0; r < 8; ++r)
      G[(size_t)(m0 + mb + r) * G4H + n0 + j * 16 + nn] = acc[j][r];
}

// ---------------------------------------------------------------------------
// Kernel 3: LSTM recurrence + output linear.
// 2 WGs x 256 thr; WG g owns batch rows [16g, 16g+16). h,c live in LDS (f32);
// recurrent GEMM (16 x 1024 x 256) per step via bf16 WMMA.
// ---------------------------------------------------------------------------
__global__ __launch_bounds__(256) void lstm_kernel(
    const float* __restrict__ G, const __bf16* __restrict__ Whhb,
    const float* __restrict__ b_ih, const float* __restrict__ b_hh,
    const float* __restrict__ Wo, const float* __restrict__ bo,
    float* __restrict__ out) {
  extern __shared__ char smem[];
  __bf16* hb = (__bf16*)smem;                    //  8192 B : h in bf16
  float*  hf = (float*)(smem + 8192);            // 16384 B : h f32
  float*  cf = hf + 16 * H_;                     // 16384 B : c f32
  float*  sG = cf + 16 * H_;                     // 65536 B : gates f32
  // total: 106,496 B

  const int g    = blockIdx.x;                   // batch group (0..1)
  const int tid  = threadIdx.x;
  const int lane = tid & 31;
  const int wave = tid >> 5;

  for (int k = tid; k < 16 * H_; k += 256) { hf[k] = 0.f; cf[k] = 0.f; }
  __syncthreads();

  for (int t = 0; t < T_; ++t) {
    // h -> bf16 for the WMMA A-operand
    for (int k = tid; k < 16 * H_; k += 256) hb[k] = (__bf16)hf[k];
    __syncthreads();

    // gates_pre = h @ W_hh^T  (M=16, N=1024, K=256)
    for (int nt = wave; nt < 64; nt += 8) {
      v8f acc = {};
#pragma unroll
      for (int kk = 0; kk < 8; ++kk) {
        v16bf A  = load_a_rm(hb + kk * 32, H_);
        v16bf Bv = load_b_nk(Whhb, H_, nt * 16, kk * 32); // B[k][n]=W_hh[n][k]
        acc = __builtin_amdgcn_wmma_f32_16x16x32_bf16(
            false, A, false, Bv, (short)0, acc, false, false);
      }
      const int nn   = nt * 16 + (lane & 15);    // gate unit u (0..1023)
      const int mbv  = (lane >> 4) * 8;
      const float bs = b_ih[nn] + b_hh[nn];
#pragma unroll
      for (int r = 0; r < 8; ++r) {
        const int mrow = mbv + r;                // batch row within WG
        const int brow = g * 16 + mrow;          // global batch index
        sG[mrow * G4H + nn] =
            acc[r] + G[(size_t)(brow * T_ + t) * G4H + nn] + bs;
      }
    }
    __syncthreads();

    // pointwise LSTM update (f32), gate order i,f,g,o
    for (int k = tid; k < 16 * H_; k += 256) {
      const int mrow = k >> 8, q = k & (H_ - 1);
      const float ig = sG[mrow * G4H + q];
      const float fg = sG[mrow * G4H + H_ + q];
      const float gg = sG[mrow * G4H + 2 * H_ + q];
      const float og = sG[mrow * G4H + 3 * H_ + q];
      const float c  = sigf(fg) * cf[k] + sigf(ig) * tanhf(gg);
      cf[k] = c;
      hf[k] = sigf(og) * tanhf(c);
    }
    __syncthreads();
  }

  // out[b][n] = h_last[b] . Wo[n] + bo[n]  (f32 VALU; tiny)
  {
    const int n = tid;                           // 0..255
    const float* wr = Wo + (size_t)n * H_;
    for (int r = 0; r < 16; ++r) {
      float acc = 0.f;
      for (int k = 0; k < H_; ++k) acc += hf[r * H_ + k] * wr[k];
      out[(size_t)(g * 16 + r) * N_ + n] = acc + bo[n];
    }
  }
}

// ---------------------------------------------------------------------------
// Host launcher
// ---------------------------------------------------------------------------
extern "C" void kernel_launch(void* const* d_in, const int* in_sizes, int n_in,
                              void* d_out, int out_size, void* d_ws,
                              size_t ws_size, hipStream_t stream) {
  (void)in_sizes; (void)n_in; (void)out_size; (void)ws_size;
  const float* x     = (const float*)d_in[0];
  const float* adj   = (const float*)d_in[1];
  const float* Wg    = (const float*)d_in[2];
  const float* a_src = (const float*)d_in[3];
  const float* a_dst = (const float*)d_in[4];
  const float* gat_b = (const float*)d_in[5];
  const float* W_ih  = (const float*)d_in[6];
  const float* W_hh  = (const float*)d_in[7];
  const float* b_ih  = (const float*)d_in[8];
  const float* b_hh  = (const float*)d_in[9];
  const float* Wo    = (const float*)d_in[10];
  const float* bo    = (const float*)d_in[11];
  float* out = (float*)d_out;

  // Workspace carve (~37 MB)
  char* ws = (char*)d_ws;
  __bf16* Xb   = (__bf16*)ws;                              // 16 MB  (1024x8192)
  __bf16* Wihb = (__bf16*)(ws + (size_t)16 * 1024 * 1024); // 16 MB  (1024x8192)
  __bf16* Whhb = (__bf16*)(ws + (size_t)32 * 1024 * 1024); // 512 KB (1024x256)
  float*  G    = (float*)(ws + (size_t)33 * 1024 * 1024);  // 4 MB   (1024x1024)

  const int SMEM_GAT  = 154880;
  const int SMEM_LSTM = 106496;
  (void)hipFuncSetAttribute((const void*)gat_kernel,
                            hipFuncAttributeMaxDynamicSharedMemorySize,
                            SMEM_GAT);
  (void)hipFuncSetAttribute((const void*)lstm_kernel,
                            hipFuncAttributeMaxDynamicSharedMemorySize,
                            SMEM_LSTM);

  // Weight conversions (bf16 for the tensor-core GEMMs)
  {
    const int nih = G4H * IN_;   // 8,388,608
    f32_to_bf16_kernel<<<(nih + 255) / 256, 256, 0, stream>>>(W_ih, Wihb, nih);
    const int nhh = G4H * H_;    // 262,144
    f32_to_bf16_kernel<<<(nhh + 255) / 256, 256, 0, stream>>>(W_hh, Whhb, nhh);
  }

  // 1) GAT: one WG per (b,t)
  gat_kernel<<<B_ * T_, 256, SMEM_GAT, stream>>>(x, adj, Wg, a_src, a_dst,
                                                 gat_b, Xb);
  // 2) LSTM input projection: G = X @ W_ih^T
  lstm_in_gemm<<<dim3(G4H / 64, (B_ * T_) / 128), 256, 0, stream>>>(Xb, Wihb, G);
  // 3) LSTM recurrence + output linear
  lstm_kernel<<<2, 256, SMEM_LSTM, stream>>>(G, Whhb, b_ih, b_hh, Wo, bo, out);
}
